// FM_32908039422140
// MI455X (gfx1250) — matrix operand order repaired and checked by reading, not compiled, not run on hardware
//
#include <hip/hip_runtime.h>
#include <hip/hip_bf16.h>

typedef __attribute__((ext_vector_type(16))) _Float16 v16h;
typedef __attribute__((ext_vector_type(8)))  float    v8f;
typedef __attribute__((ext_vector_type(2)))  float    f32x2;
typedef __attribute__((ext_vector_type(4)))  float    f32x4;

namespace {
constexpr int kD     = 13;       // dense features
constexpr int kS     = 26;       // sparse fields
constexpr int kFeat  = 100000;   // one-hot depth
constexpr int kK     = 64;       // FM latent dim
constexpr int kTile  = 16;       // samples per wave (WMMA M)
constexpr int kWaves = 8;        // waves per block
constexpr int kStride = 68;      // padded LDS row stride (floats) -> conflict-free reduce
}

// Dense micro-GEMM term via V_WMMA_F32_16X16X32_F16, fully specialized so the
// two passes (dense@V and dense^2@V^2) are straight-line code with no selects.
template <bool kSquare>
__device__ __forceinline__ void dense_term(const v16h a,
                                           const float* __restrict__ V,
                                           float (&stage)[kTile][kStride],
                                           const int lane, const int m,
                                           f32x2 (&acc)[kTile])
{
#pragma unroll
    for (int c = 0; c < 4; ++c) {
        // B fragment 32x16: lanes hold N=lane&15, K=8*(lane/16)+half-index in halves 0..7;
        // halves 8..15 (K=16..31) and K>=13 are zero here (K_actual=13).
        v16h bf = {};
        {
            const int n = c * 16 + m;
            const int kBase = (lane < 16) ? 0 : 8;
#pragma unroll
            for (int e = 0; e < 8; ++e) {
                const int k = kBase + e;
                if (k < kD) {
                    const float vv = V[k * kK + n];
                    bf[e] = (_Float16)(kSquare ? vv * vv : vv);
                }
            }
        }
        v8f c8 = {};
        c8 = __builtin_amdgcn_wmma_f32_16x16x32_f16(
            false, a, false, bf, (short)0, c8, false, false);
        // C layout: vgpr r -> M = r (+8 for lanes 16-31), N = (lane&15) + 16c
        const int n = c * 16 + m;
        const int mBase = (lane < 16) ? 0 : 8;
#pragma unroll
        for (int r = 0; r < 8; ++r)
            stage[mBase + r][n] = c8[r];
    }
    // LDS round-trip converts C-fragment layout -> gather layout (lane owns
    // k = 2*lane, 2*lane+1). Same wave, LDS in-order: no barrier needed.
#pragma unroll
    for (int mm = 0; mm < kTile; ++mm)
        acc[mm] = *(const f32x2*)&stage[mm][2 * lane];
}

__global__ __launch_bounds__(kWaves * 32)
void fm_fused_kernel(const float* __restrict__ dense,
                     const int*   __restrict__ sparse,
                     const float* __restrict__ w,
                     const float* __restrict__ V,
                     const float* __restrict__ bias,
                     float* __restrict__ out)
{
    __shared__ float sStage[kWaves][kTile][kStride]; // 34.8 KB: WMMA C staging + reduce
    __shared__ int   sIdx  [kWaves][kTile][kS];      // 13 KB: global row ids (offset folded)
    __shared__ float sDense[kWaves][kTile][kD];      // 6.5 KB
    __shared__ float sLin  [kWaves][kTile][2];       // 1 KB: linear-term partials

    const int wave = threadIdx.x >> 5;
    const int lane = threadIdx.x & 31;
    const int m    = lane & 15;
    const int tileBase = (blockIdx.x * kWaves + wave) * kTile;

    // ---- stage dense rows and sparse row-ids (one-hot offsets folded in) ----
    float* dRow = &sDense[wave][0][0];
    for (int i = lane; i < kTile * kD; i += 32)
        dRow[i] = dense[(size_t)tileBase * kD + i];
    int* iRow = &sIdx[wave][0][0];
    for (int i = lane; i < kTile * kS; i += 32) {
        const int s = i % kS;
        iRow[i] = sparse[(size_t)tileBase * kS + i] + (kD + s * kFeat);
    }

    // ---- A fragments: f16 16x32 layout, K zero-padded beyond 13 ----
    // lanes 0-15 hold M=lane, K=0..7 in halves 0..7 (halves 8..15 = K=16..23 -> 0)
    // lanes 16-31 hold M=lane-16, K=8..15 in halves 0..7 (halves 8..15 = K=24..31 -> 0)
    v16h a = {}, a2 = {};
    {
        const int kBase = (lane < 16) ? 0 : 8;
#pragma unroll
        for (int e = 0; e < 8; ++e) {
            const int k = kBase + e;
            const float x = (k < kD) ? sDense[wave][m][k] : 0.0f;
            a[e]  = (_Float16)x;
            a2[e] = (_Float16)(x * x);
        }
    }

    f32x2 xv[kTile];   // lane owns k = 2*lane, 2*lane+1 for all 16 samples
    f32x2 x2[kTile];
    dense_term<false>(a,  V, sStage[wave], lane, m, xv);  // dense   @ V
    dense_term<true >(a2, V, sStage[wave], lane, m, x2);  // dense^2 @ V^2

    // ---- hot loop: 26 fields x 16 samples; wave covers a 256B row as 32xb64.
    // V gathers are non-temporal: the 665MB table has ~16% row reuse and must
    // not evict the L2-resident 10MB w table. ----
#pragma unroll 1
    for (int s = 0; s < kS; ++s) {
        int rows[kTile];
#pragma unroll
        for (int mm = 0; mm < kTile; ++mm) rows[mm] = sIdx[wave][mm][s];
#pragma unroll
        for (int mm = 0; mm < kTile; ++mm) {
            const f32x2 v = __builtin_nontemporal_load(
                (const f32x2*)(V + (size_t)rows[mm] * kK + 2 * lane));
            xv[mm].x += v.x;
            xv[mm].y += v.y;
            x2[mm].x = fmaf(v.x, v.x, x2[mm].x);
            x2[mm].y = fmaf(v.y, v.y, x2[mm].y);
        }
    }

    // ---- first-order (linear) term: lanes<16 take s=0..12, lanes>=16 s=13..25 ----
    {
        float linp = 0.0f;
        const int s0 = (lane < 16) ? 0 : 13;
#pragma unroll
        for (int ss = 0; ss < 13; ++ss)
            linp += w[(size_t)sIdx[wave][m][s0 + ss]];
        if (lane < 16) {
#pragma unroll
            for (int d = 0; d < kD; ++d)
                linp = fmaf(sDense[wave][m][d], w[d], linp);
            linp += bias[0];
        }
        sLin[wave][m][(lane < 16) ? 0 : 1] = linp;
    }

    // ---- second-order reduction: 0.5 * sum_k (xv^2 - x2v2) ----
#pragma unroll
    for (int mm = 0; mm < kTile; ++mm) {
        const float pm = (xv[mm].x * xv[mm].x - x2[mm].x)
                       + (xv[mm].y * xv[mm].y - x2[mm].y);
        sStage[wave][mm][lane] = pm;
    }
    if (lane < 16) {
        float tot = 0.0f;
#pragma unroll
        for (int i = 0; i < 32; i += 4) {
            const f32x4 q = *(const f32x4*)&sStage[wave][m][i];
            tot += q.x + q.y + q.z + q.w;
        }
        const float z = sLin[wave][m][0] + sLin[wave][m][1] + 0.5f * tot;
        out[tileBase + m] = 1.0f / (1.0f + __expf(-z));
    }
}

extern "C" void kernel_launch(void* const* d_in, const int* in_sizes, int n_in,
                              void* d_out, int out_size, void* d_ws, size_t ws_size,
                              hipStream_t stream) {
    (void)n_in; (void)out_size; (void)d_ws; (void)ws_size;
    const float* dense  = (const float*)d_in[0];  // [B,13] f32
    const int*   sparse = (const int*)d_in[1];    // [B,26] i32
    const float* w      = (const float*)d_in[2];  // [TOTAL,1] f32
    const float* V      = (const float*)d_in[3];  // [TOTAL,64] f32
    const float* bias   = (const float*)d_in[4];  // [1] f32
    float* out = (float*)d_out;                   // [B,1] f32

    const int samples = in_sizes[0] / kD;               // B = 16384
    const int blocks  = samples / (kTile * kWaves);     // 128 blocks x 256 threads
    fm_fused_kernel<<<blocks, kWaves * 32, 0, stream>>>(dense, sparse, w, V, bias, out);
}
